// MSDeformAttnPixelDecoder_78142634983523
// MI455X (gfx1250) — matrix operand order repaired and checked by reading, not compiled, not run on hardware
//
#include <hip/hip_runtime.h>

// ---------------- problem constants ----------------
#define BATCH   2
#define DMODEL  256
#define NHEAD   8
#define DHEAD   32
#define NPOINT  4
#define NLVL    3
#define DFF     1024
#define NLAYERS 6
#define LQTOT   12096            // 96*96 + 48*48 + 24*24
#define MROWS   (BATCH * LQTOT)  // 24192 = 189 * 128

__device__ __constant__ int c_HL[NLVL]     = {96, 48, 24};
__device__ __constant__ int c_WL[NLVL]     = {96, 48, 24};
__device__ __constant__ int c_STARTS[NLVL] = {0, 9216, 11520};

typedef _Float16 h8v  __attribute__((ext_vector_type(8)));
typedef _Float16 h16v __attribute__((ext_vector_type(16)));
typedef float    f8v  __attribute__((ext_vector_type(8)));

// ---------------- Tensor Data Mover support (gfx1250) ----------------
#if defined(__AMDGCN__) && __has_builtin(__builtin_amdgcn_tensor_load_to_lds)
#define USE_TDM 1
#else
#define USE_TDM 0
#endif

#if USE_TDM
typedef unsigned int u32x4 __attribute__((ext_vector_type(4)));
typedef int          i32x4 __attribute__((ext_vector_type(4)));
typedef int          i32x8 __attribute__((ext_vector_type(8)));

// Load a [64][32] f16 tile (rows of 32 contiguous halves) from a [N][K] f16
// tensor at row n0, col k0 into LDS at lds_off. Rows beyond N are zero-filled
// by TDM OOB handling (tensor_dim1 = N).
__device__ __forceinline__ void tdm_load_b_tile(const _Float16* tile_start,
                                                unsigned lds_off, int K, int N) {
    unsigned long long ga = (unsigned long long)tile_start;
    unsigned d0 = (unsigned)K;   // tensor_dim0 (contiguous dim, elements)
    unsigned d1 = (unsigned)N;   // tensor_dim1 (rows)
    u32x4 g0;
    g0[0] = 1u;                                               // count=1, user mode
    g0[1] = lds_off;                                          // lds_addr (bytes)
    g0[2] = (unsigned)ga;                                     // global_addr[31:0]
    g0[3] = (unsigned)((ga >> 32) & 0x01FFFFFFu) | (2u << 30);// addr[56:32], type=2
    i32x8 g1;
    g1[0] = (int)(1u << 16);                                  // data_size=1 (2 bytes)
    g1[1] = (int)((d0 & 0xFFFFu) << 16);                      // tensor_dim0[15:0] @ bit48
    g1[2] = (int)(((d0 >> 16) & 0xFFFFu) | ((d1 & 0xFFFFu) << 16));
    g1[3] = (int)(((d1 >> 16) & 0xFFFFu) | (32u << 16));      // tile_dim0 = 32
    g1[4] = 64;                                               // tile_dim1 = 64, tile_dim2=0
    g1[5] = (int)(unsigned)K;                                 // tensor_dim0_stride lo32
    g1[6] = 0;                                                // stride0 hi16 | stride1 lo16
    g1[7] = 0;
    i32x4 z4 = {0, 0, 0, 0};
    i32x8 z8 = {0, 0, 0, 0, 0, 0, 0, 0};
    __builtin_amdgcn_tensor_load_to_lds(g0, g1, z4, z4, z8, 0);
}

__device__ __forceinline__ void wait_tensorcnt0() {
#if __has_builtin(__builtin_amdgcn_s_wait_tensorcnt)
    __builtin_amdgcn_s_wait_tensorcnt(0);
#else
    asm volatile("s_wait_tensorcnt 0x0" ::: "memory");
#endif
}
#endif  // USE_TDM

// ---------------- weight transpose-convert: [L][K][N] f32 -> [L][N][K] f16 ----
// grid: (K/32, N/32, L), block: (32, 8)
__global__ __launch_bounds__(256)
void cvtT_kernel(const float* __restrict__ in, _Float16* __restrict__ out,
                 int K, int N) {
    __shared__ float t[32][33];
    int l  = blockIdx.z;
    int k0 = blockIdx.x * 32;
    int n0 = blockIdx.y * 32;
    const float* ip = in + (size_t)l * K * N;
    _Float16* op    = out + (size_t)l * N * K;
#pragma unroll
    for (int i = 0; i < 4; ++i) {
        int k = k0 + threadIdx.y + i * 8;
        t[threadIdx.y + i * 8][threadIdx.x] = ip[(size_t)k * N + n0 + threadIdx.x];
    }
    __syncthreads();
#pragma unroll
    for (int i = 0; i < 4; ++i) {
        int n = n0 + threadIdx.y + i * 8;
        op[(size_t)n * K + k0 + threadIdx.x] = (_Float16)t[threadIdx.x][threadIdx.y + i * 8];
    }
}

// ---------------- flatten_permute: [B,D,HW] -> [B, start+hw, D] (+level embed)
// grid: (HW/32, D/32, B), block: (32, 8)
__global__ __launch_bounds__(256)
void prep_kernel(const float* __restrict__ srcL, const float* __restrict__ posL,
                 const float* __restrict__ lvlEmb,
                 float* __restrict__ srcT, float* __restrict__ posT,
                 int HW, int start) {
    __shared__ float ts[32][33];
    __shared__ float tp[32][33];
    int b   = blockIdx.z;
    int hw0 = blockIdx.x * 32;
    int d0  = blockIdx.y * 32;
    const float* sp = srcL + (size_t)b * DMODEL * HW;
    const float* pp = posL + (size_t)b * DMODEL * HW;
#pragma unroll
    for (int i = 0; i < 4; ++i) {
        int d = d0 + threadIdx.y + i * 8;
        ts[threadIdx.y + i * 8][threadIdx.x] = sp[(size_t)d * HW + hw0 + threadIdx.x];
        tp[threadIdx.y + i * 8][threadIdx.x] = pp[(size_t)d * HW + hw0 + threadIdx.x];
    }
    __syncthreads();
#pragma unroll
    for (int i = 0; i < 4; ++i) {
        int hw = hw0 + threadIdx.y + i * 8;
        int d  = d0 + threadIdx.x;
        size_t orow = (size_t)b * LQTOT + start + hw;
        srcT[orow * DMODEL + d] = ts[threadIdx.x][threadIdx.y + i * 8];
        posT[orow * DMODEL + d] = tp[threadIdx.x][threadIdx.y + i * 8] + lvlEmb[d];
    }
}

// ---------------- WMMA GEMM: C[M,N] = (A (+A2)) [M,K] * B^T[N,K] + bias -------
// A,A2 f32 row-major [M,K]; BwT f16 row-major [N,K] (pre-transposed weights).
// Tile 128x64, BK=32. 256 thr = 8 waves, each wave: 16 rows x 64 cols
// = 4x v_wmma_f32_16x16x32_f16 accumulators. B tile fetched by the TDM
// (tensor_load_to_lds) when available; A tile staged with fused q=src+pos add
// and f32->f16 conversion.
__global__ __launch_bounds__(256)
void gemm_kernel(const float* __restrict__ A, const float* __restrict__ A2,
                 const _Float16* __restrict__ BwT, const float* __restrict__ bias,
                 float* __restrict__ C, int M, int N, int K, int relu) {
    __shared__ _Float16 Ah[128 * 32];  // [row][k]
    __shared__ _Float16 Bh[64 * 32];   // [n][k]
    const int tid  = threadIdx.x;
    const int wave = tid >> 5;
    const int lane = tid & 31;
    const int m0 = blockIdx.y * 128;
    const int n0 = blockIdx.x * 64;

#if USE_TDM
    const unsigned bh_off = (unsigned)(unsigned long long)(&Bh[0]);
#endif

    f8v acc[4] = {};

    for (int k0 = 0; k0 < K; k0 += 32) {
#if USE_TDM
        if (wave == 0)
            tdm_load_b_tile(BwT + (size_t)n0 * K + k0, bh_off, K, N);
#endif
        // A tile: 128x32 f32 -> f16 (fused q = src + pos when A2 != null)
#pragma unroll
        for (int i = 0; i < 4; ++i) {
            int idx = tid + i * 256;     // 0..1023 float4 chunks
            int row = idx >> 3;
            int c4  = (idx & 7) << 2;
            const float4* ap = (const float4*)(A + (size_t)(m0 + row) * K + k0 + c4);
            float4 v = *ap;
            if (A2) {
                const float4* qp = (const float4*)(A2 + (size_t)(m0 + row) * K + k0 + c4);
                float4 w = *qp;
                v.x += w.x; v.y += w.y; v.z += w.z; v.w += w.w;
            }
            _Float16* dst = &Ah[row * 32 + c4];
            dst[0] = (_Float16)v.x; dst[1] = (_Float16)v.y;
            dst[2] = (_Float16)v.z; dst[3] = (_Float16)v.w;
        }
#if !USE_TDM
        // fallback B tile loader: contiguous 16B chunks from [N][K]
        {
            int n  = tid >> 2;           // 0..63
            int kc = (tid & 3) << 3;     // 0,8,16,24
            h8v v = {};
            if (n0 + n < N)
                v = *(const h8v*)(BwT + (size_t)(n0 + n) * K + k0 + kc);
            *(h8v*)&Bh[n * 32 + kc] = v;
        }
#endif
#if USE_TDM
        if (wave == 0) wait_tensorcnt0();
#endif
        __syncthreads();

        const int r  = lane & 15;
        const int hi = lane >> 4;
        // A fragment (16-bit A 16x32 layout): lane row = wave*16 + r,
        // halves [hi*8 .. hi*8+7] then [16+hi*8 .. 16+hi*8+7]
        h8v alo = *(const h8v*)&Ah[(wave * 16 + r) * 32 + hi * 8];
        h8v ahi = *(const h8v*)&Ah[(wave * 16 + r) * 32 + 16 + hi * 8];
        h16v afrag;
#pragma unroll
        for (int j = 0; j < 8; ++j) { afrag[j] = alo[j]; afrag[8 + j] = ahi[j]; }

#pragma unroll
        for (int nt = 0; nt < 4; ++nt) {
            // B fragment (16-bit B 32x16 layout): lane col = nt*16 + r,
            // halves [hi*8 .. hi*8+7] then [16+hi*8 .. 16+hi*8+7] (matches A)
            h8v blo = *(const h8v*)&Bh[(nt * 16 + r) * 32 + hi * 8];
            h8v bhi = *(const h8v*)&Bh[(nt * 16 + r) * 32 + 16 + hi * 8];
            h16v bfrag;
#pragma unroll
            for (int j = 0; j < 8; ++j) { bfrag[j] = blo[j]; bfrag[8 + j] = bhi[j]; }
            acc[nt] = __builtin_amdgcn_wmma_f32_16x16x32_f16(
                false, afrag, false, bfrag, (short)0, acc[nt], false, false);
        }
        __syncthreads();
    }

    // store: C/D f32 layout -> row = wave*16 + hi*8 + vr, col = nt*16 + (lane&15)
    const int r  = lane & 15;
    const int hi = lane >> 4;
#pragma unroll
    for (int nt = 0; nt < 4; ++nt) {
        int col = n0 + nt * 16 + r;
        if (col < N) {
            float bv = bias ? bias[col] : 0.f;
#pragma unroll
            for (int vr = 0; vr < 8; ++vr) {
                int row = m0 + wave * 16 + hi * 8 + vr;
                float o = acc[nt][vr] + bv;
                if (relu) o = fmaxf(o, 0.f);
                C[(size_t)row * N + col] = o;
            }
        }
    }
}

// ---------------- MSDeformAttn core (fused softmax + bilinear gather) ----------
// one block per query m, one wave per head, lane = channel (DH=32)
__global__ __launch_bounds__(256)
void msdeform_kernel(const float* __restrict__ value, const float* __restrict__ off,
                     const float* __restrict__ aw, float* __restrict__ out) {
    int m    = blockIdx.x;
    int h    = threadIdx.x >> 5;
    int lane = threadIdx.x & 31;
    int b  = m / LQTOT;
    int lq = m - b * LQTOT;

    // reference point (same for all target levels; based on query's own level)
    float rx, ry;
    {
        int l, loc;
        if (lq < 9216)       { l = 0; loc = lq; }
        else if (lq < 11520) { l = 1; loc = lq - 9216; }
        else                 { l = 2; loc = lq - 11520; }
        int Wl = c_WL[l];
        int y = loc / Wl, x = loc - y * Wl;
        rx = (x + 0.5f) / (float)Wl;
        ry = (y + 0.5f) / (float)c_HL[l];
    }

    // softmax over LVL*P = 12 logits for this head
    const float* lg = aw + (size_t)m * (NHEAD * NLVL * NPOINT) + h * (NLVL * NPOINT);
    float mx = -1e30f;
#pragma unroll
    for (int j = 0; j < 12; ++j) mx = fmaxf(mx, lg[j]);
    float ex[12], s = 0.f;
#pragma unroll
    for (int j = 0; j < 12; ++j) { ex[j] = expf(lg[j] - mx); s += ex[j]; }
    float inv = 1.f / s;

    const float* op    = off + (size_t)m * (NHEAD * NLVL * NPOINT * 2) + h * (NLVL * NPOINT * 2);
    const float* vbase = value + (size_t)b * LQTOT * DMODEL + h * DHEAD + lane;
    float accv = 0.f;

#pragma unroll
    for (int l = 0; l < NLVL; ++l) {
        int Wl = c_WL[l], Hl = c_HL[l];
        const float* vl = vbase + (size_t)c_STARTS[l] * DMODEL;
#pragma unroll
        for (int p = 0; p < NPOINT; ++p) {
            float ox = op[l * 8 + p * 2 + 0];
            float oy = op[l * 8 + p * 2 + 1];
            float w  = ex[l * NPOINT + p] * inv;
            // loc = ref + off/(W,H); px = loc.x*W - 0.5  ==  rx*W + ox - 0.5
            float px = rx * (float)Wl + ox - 0.5f;
            float py = ry * (float)Hl + oy - 0.5f;
            float x0f = floorf(px), y0f = floorf(py);
            float fx = px - x0f, fy = py - y0f;
            int x0 = (int)x0f, y0 = (int)y0f;
#pragma unroll
            for (int dy = 0; dy < 2; ++dy) {
                int yi = y0 + dy;
                float wy = dy ? fy : (1.f - fy);
                bool vy = (yi >= 0) && (yi < Hl);
                int yc = yi < 0 ? 0 : (yi > Hl - 1 ? Hl - 1 : yi);
#pragma unroll
                for (int dx = 0; dx < 2; ++dx) {
                    int xi = x0 + dx;
                    float wx = dx ? fx : (1.f - fx);
                    bool vx = (xi >= 0) && (xi < Wl);
                    int xc = xi < 0 ? 0 : (xi > Wl - 1 ? Wl - 1 : xi);
                    float cw = (vx && vy) ? (w * wx * wy) : 0.f;
                    float v = vl[(size_t)(yc * Wl + xc) * DMODEL];  // coalesced 128B/wave
                    accv = fmaf(cw, v, accv);
                }
            }
        }
    }
    out[(size_t)m * DMODEL + h * DHEAD + lane] = accv;
}

// ---------------- residual add + LayerNorm: Y = LN(X + R) * g + be ------------
// one wave per row; 8 rows per block
__global__ __launch_bounds__(256)
void ln_kernel(const float* __restrict__ X, const float* __restrict__ R,
               const float* __restrict__ g, const float* __restrict__ be,
               float* __restrict__ Y) {
    int row  = blockIdx.x * 8 + (threadIdx.x >> 5);
    int lane = threadIdx.x & 31;
    const float* x = X + (size_t)row * DMODEL;
    const float* r = R + (size_t)row * DMODEL;
    float v[8], s = 0.f, s2 = 0.f;
#pragma unroll
    for (int i = 0; i < 8; ++i) {
        int c = i * 32 + lane;
        float t = x[c] + r[c];
        v[i] = t; s += t; s2 += t * t;
    }
#pragma unroll
    for (int o = 16; o > 0; o >>= 1) {
        s  += __shfl_xor(s, o, 32);
        s2 += __shfl_xor(s2, o, 32);
    }
    float mu  = s * (1.f / DMODEL);
    float var = s2 * (1.f / DMODEL) - mu * mu;
    float rs  = rsqrtf(var + 1e-5f);
    float* y = Y + (size_t)row * DMODEL;
#pragma unroll
    for (int i = 0; i < 8; ++i) {
        int c = i * 32 + lane;
        y[c] = (v[i] - mu) * rs * g[c] + be[c];
    }
}

// ---------------- host orchestration ----------------
static void run_gemm(const float* A, const float* A2, const _Float16* BwT,
                     const float* bias, float* C, int M, int N, int K, int relu,
                     hipStream_t stream) {
    dim3 grid((N + 63) / 64, M / 128);
    gemm_kernel<<<grid, 256, 0, stream>>>(A, A2, BwT, bias, C, M, N, K, relu);
}

extern "C" void kernel_launch(void* const* d_in, const int* in_sizes, int n_in,
                              void* d_out, int out_size, void* d_ws, size_t ws_size,
                              hipStream_t stream) {
    (void)in_sizes; (void)n_in; (void)out_size; (void)ws_size;
    // inputs (setup_inputs dict order)
    const float* src_l[3] = {(const float*)d_in[0], (const float*)d_in[2], (const float*)d_in[4]};
    const float* pos_l[3] = {(const float*)d_in[1], (const float*)d_in[3], (const float*)d_in[5]};
    const float* level_embed = (const float*)d_in[6];
    const float* Wv   = (const float*)d_in[7];
    const float* bv   = (const float*)d_in[8];
    const float* Woff = (const float*)d_in[9];
    const float* boff = (const float*)d_in[10];
    const float* Waw  = (const float*)d_in[11];
    const float* baw  = (const float*)d_in[12];
    const float* Wout = (const float*)d_in[13];
    const float* bout = (const float*)d_in[14];
    const float* g1   = (const float*)d_in[15];
    const float* be1  = (const float*)d_in[16];
    const float* W1   = (const float*)d_in[17];
    const float* b1   = (const float*)d_in[18];
    const float* W2   = (const float*)d_in[19];
    const float* b2   = (const float*)d_in[20];
    const float* g2   = (const float*)d_in[21];
    const float* be2  = (const float*)d_in[22];

    const size_t Mr = MROWS;
    // workspace partition
    float* f = (float*)d_ws;
    float* posT = f; f += Mr * DMODEL;
    float* act  = f; f += Mr * DMODEL;
    float* val  = f; f += Mr * DMODEL;
    float* core = f; f += Mr * DMODEL;
    float* lno  = f; f += Mr * DMODEL;
    float* offb = f; f += Mr * (NHEAD * NLVL * NPOINT * 2);  // 192
    float* awb  = f; f += Mr * (NHEAD * NLVL * NPOINT);      // 96
    float* ff1  = f; f += Mr * DFF;
    _Float16* hw = (_Float16*)f;
    const size_t szWv = (size_t)NLAYERS * DMODEL * DMODEL;
    const size_t szWo = (size_t)NLAYERS * DMODEL * 192;
    const size_t szWa = (size_t)NLAYERS * DMODEL * 96;
    const size_t szW1 = (size_t)NLAYERS * DMODEL * DFF;
    const size_t szW2 = (size_t)NLAYERS * DFF * DMODEL;
    _Float16* Wv16   = hw; hw += szWv;   // [L][N=256][K=256]
    _Float16* Woff16 = hw; hw += szWo;   // [L][N=192][K=256]
    _Float16* Waw16  = hw; hw += szWa;   // [L][N= 96][K=256]
    _Float16* Wout16 = hw; hw += szWv;   // [L][N=256][K=256]
    _Float16* W116   = hw; hw += szW1;   // [L][N=1024][K=256]
    _Float16* W216   = hw; hw += szW2;   // [L][N=256][K=1024]

    // transpose-convert weights to f16 [N][K] (all layers at once)
    auto cvtT = [&](const float* in, _Float16* out, int K, int N) {
        dim3 grid(K / 32, N / 32, NLAYERS);
        dim3 block(32, 8);
        cvtT_kernel<<<grid, block, 0, stream>>>(in, out, K, N);
    };
    cvtT(Wv,   Wv16,   DMODEL, DMODEL);
    cvtT(Woff, Woff16, DMODEL, 192);
    cvtT(Waw,  Waw16,  DMODEL, 96);
    cvtT(Wout, Wout16, DMODEL, DMODEL);
    cvtT(W1,   W116,   DMODEL, DFF);
    cvtT(W2,   W216,   DFF,    DMODEL);

    // assemble src/pos in [B, LQ, D]
    const int HWs[3]    = {9216, 2304, 576};
    const int startL[3] = {0, 9216, 11520};
    for (int l = 0; l < 3; ++l) {
        dim3 grid(HWs[l] / 32, DMODEL / 32, BATCH);
        dim3 block(32, 8);
        prep_kernel<<<grid, block, 0, stream>>>(src_l[l], pos_l[l],
                                                level_embed + (size_t)l * DMODEL,
                                                act, posT, HWs[l], startL[l]);
    }

    for (int i = 0; i < NLAYERS; ++i) {
        // value = src @ Wv + bv
        run_gemm(act, nullptr, Wv16 + (size_t)i * DMODEL * DMODEL,
                 bv + (size_t)i * DMODEL, val, MROWS, DMODEL, DMODEL, 0, stream);
        // off = (src+pos) @ Woff + boff
        run_gemm(act, posT, Woff16 + (size_t)i * DMODEL * 192,
                 boff + (size_t)i * 192, offb, MROWS, 192, DMODEL, 0, stream);
        // aw logits = (src+pos) @ Waw + baw
        run_gemm(act, posT, Waw16 + (size_t)i * DMODEL * 96,
                 baw + (size_t)i * 96, awb, MROWS, 96, DMODEL, 0, stream);
        // deformable attention core (fused softmax)
        msdeform_kernel<<<MROWS, 256, 0, stream>>>(val, offb, awb, core);
        // attn = core @ Wout + bout   (val is free now, reuse as output)
        run_gemm(core, nullptr, Wout16 + (size_t)i * DMODEL * DMODEL,
                 bout + (size_t)i * DMODEL, val, MROWS, DMODEL, DMODEL, 0, stream);
        // src = LN(src + attn)
        ln_kernel<<<MROWS / 8, 256, 0, stream>>>(val, act,
                                                 g1 + (size_t)i * DMODEL,
                                                 be1 + (size_t)i * DMODEL, lno);
        // ff1 = relu(src @ W1 + b1)
        run_gemm(lno, nullptr, W116 + (size_t)i * DMODEL * DFF,
                 b1 + (size_t)i * DFF, ff1, MROWS, DFF, DMODEL, 1, stream);
        // ff2 = ff1 @ W2 + b2   (reuse val)
        run_gemm(ff1, nullptr, W216 + (size_t)i * DFF * DMODEL,
                 b2 + (size_t)i * DMODEL, val, MROWS, DMODEL, DFF, 0, stream);
        // src = LN(src + ff2)
        ln_kernel<<<MROWS / 8, 256, 0, stream>>>(val, lno,
                                                 g2 + (size_t)i * DMODEL,
                                                 be2 + (size_t)i * DMODEL, act);
    }

    (void)hipMemcpyAsync(d_out, act, (size_t)MROWS * DMODEL * sizeof(float),
                         hipMemcpyDeviceToDevice, stream);
}
